// CM_NTM_29566554866014
// MI455X (gfx1250) — matrix operand order repaired and checked by reading, not compiled, not run on hardware
//
#include <hip/hip_runtime.h>
#include <cstddef>

// ---------------------------------------------------------------------------
// NTM forward, T=4 chained layers.  Write head + memory update are dead code
// w.r.t. the output and are eliminated.  All GEMMs run on bf16 WMMA with fp32
// accumulation; weights/activations are converted to bf16 once in global
// memory.  GEMM tiles are staged to LDS by the Tensor Data Mover (double
// buffered, s_wait_tensorcnt) when the toolchain exposes the builtin.
// ---------------------------------------------------------------------------
#define CT 4
#define CE 512
#define CV 256
#define CH 512
#define CN 128
#define CM 64
#define CB 2048

#define BM 128          // block tile M
#define BN 64           // block tile N
#define BK 32           // K step
#define LROW 40         // padded LDS row pitch in bf16 elems (80B = 64B row + 16B pad)

typedef __attribute__((ext_vector_type(16))) __bf16 v16bf;
typedef __attribute__((ext_vector_type(8)))  float  v8f;
typedef unsigned int   u32x4 __attribute__((ext_vector_type(4)));
typedef int            i32x4 __attribute__((ext_vector_type(4)));
typedef int            i32x8 __attribute__((ext_vector_type(8)));
typedef unsigned int   u32x4v __attribute__((ext_vector_type(4)));
typedef unsigned short u16x4 __attribute__((ext_vector_type(4)));

#if defined(__gfx1250__) && __has_builtin(__builtin_amdgcn_tensor_load_to_lds) && \
    __has_builtin(__builtin_amdgcn_s_wait_tensorcnt)
#define USE_TDM 1
#else
#define USE_TDM 0
#endif

union Frag {
    u32x4v q[2];
    unsigned short u[16];
    v16bf v;
};

__device__ __forceinline__ unsigned short f2bf(float f) {
    unsigned int u = __float_as_uint(f);
    unsigned int r = u + 0x7FFFu + ((u >> 16) & 1u);   // round-to-nearest-even
    return (unsigned short)(r >> 16);
}
__device__ __forceinline__ float dev_sigmoid(float x) { return 1.0f / (1.0f + __expf(-x)); }
__device__ __forceinline__ float dev_softplus(float x) {
    return fmaxf(x, 0.0f) + log1pf(__expf(-fabsf(x)));
}

#if USE_TDM
// Issue one TDM 2-D tile load: bf16 tensor with row pitch k_stride elems,
// tile 32 (K) x tile_rows, LDS rows padded 64B+16B (pad_interval=3,
// pad_amount=3) to match the bank-conflict-free 80B LDS pitch.  rows_rem /
// k_rem are remaining tensor extents from the tile origin so OOB zero-fills.
__device__ __forceinline__ void tdm_tile_load(const unsigned short* gsrc, unsigned lds_off,
                                              unsigned rows_rem, unsigned k_rem,
                                              unsigned tile_rows, unsigned k_stride)
{
    const unsigned long long ga = (unsigned long long)gsrc;
    u32x4 g0;
    g0[0] = 1u;                                                 // count=1, user D#
    g0[1] = lds_off;                                            // lds_addr
    g0[2] = (unsigned)(ga & 0xFFFFFFFFu);                       // global_addr[31:0]
    g0[3] = (unsigned)((ga >> 32) & 0x01FFFFFFu) | (2u << 30);  // addr[56:32] | type=2
    i32x8 g1;
    g1[0] = (int)((1u << 16) | (1u << 20) | (3u << 22) | (3u << 25)); // 2B, pad, 64B int., 16B pad
    g1[1] = (int)((k_rem & 0xFFFFu) << 16);                     // tensor_dim0 low bits @63:48
    g1[2] = (int)((k_rem >> 16) | ((rows_rem & 0xFFFFu) << 16));
    g1[3] = (int)((rows_rem >> 16) | (32u << 16));              // tile_dim0 = 32
    g1[4] = (int)tile_rows;                                     // tile_dim1 (tile_dim2=0)
    g1[5] = (int)k_stride;                                      // tensor_dim0_stride[31:0]
    g1[6] = 0;
    g1[7] = 0;
    i32x4 z4 = {0, 0, 0, 0};
#if __clang_major__ >= 23
    i32x8 z8 = {0, 0, 0, 0, 0, 0, 0, 0};
    __builtin_amdgcn_tensor_load_to_lds(g0, g1, z4, z4, z8, 0);
#else
    __builtin_amdgcn_tensor_load_to_lds(g0, g1, z4, z4, 0);
#endif
}
#endif

// ---------------------------------------------------------------------------
// WMMA GEMM:  C[M,N] = epilogue( A[M,K](bf16) @ W[N,K](bf16)^T + bias(+bias2)
//             (+C fp32) ), C row stride ldc, fp32 or bf16 output.
// Block: 256 threads (8 wave32s), tile 128x64, BK=32.  Waves in a 4x2 grid;
// each wave owns a 32x32 tile: 2 A-frags x 2 B-frags -> 4 WMMAs per K-step.
// Requires M % 128 == 0, K % 32 == 0;  N guarded (OOB weight rows -> 0).
// epilogue: 0 none, 1 tanh, 2 sigmoid.
// ---------------------------------------------------------------------------
__global__ __launch_bounds__(256)
void gemm_bf16_wmma(const unsigned short* __restrict__ A,
                    const unsigned short* __restrict__ W,
                    const float* __restrict__ bias, const float* __restrict__ bias2,
                    void* __restrict__ Cout,
                    int Nc, int K, int ldc, int epilogue, int accumulate, int out_bf16)
{
    __shared__ unsigned short As[2][BM][LROW];   // 2 x 10240 B
    __shared__ unsigned short Bs[2][BN][LROW];   // 2 x  5120 B

    const int tid  = threadIdx.x;
    const int wave = tid >> 5;
    const int lane = tid & 31;
    const int lm   = lane & 15;
    const int halfsel = lane >> 4;

    const int block_m = blockIdx.y * BM;
    const int block_n = blockIdx.x * BN;
    const int wm = wave & 3;          // 4 waves along M (32 rows each)
    const int wn = wave >> 2;         // 2 waves along N (32 cols each)

    v8f acc00 = {0.f,0.f,0.f,0.f,0.f,0.f,0.f,0.f};
    v8f acc01 = acc00, acc10 = acc00, acc11 = acc00;

    const int ntiles = K / BK;

#if USE_TDM
    const unsigned ldsA0 = (unsigned)(unsigned long long)(const void*)&As[0][0][0];
    const unsigned ldsA1 = (unsigned)(unsigned long long)(const void*)&As[1][0][0];
    const unsigned ldsB0 = (unsigned)(unsigned long long)(const void*)&Bs[0][0][0];
    const unsigned ldsB1 = (unsigned)(unsigned long long)(const void*)&Bs[1][0][0];
    if (wave == 0) {
        tdm_tile_load(A + (size_t)block_m * K, ldsA0, (unsigned)(CB - block_m),
                      (unsigned)K, BM, (unsigned)K);
        tdm_tile_load(W + (size_t)block_n * K, ldsB0, (unsigned)(Nc - block_n),
                      (unsigned)K, BN, (unsigned)K);
    }
#endif

    for (int t = 0; t < ntiles; ++t) {
#if USE_TDM
        const int cur = t & 1;
        if (wave == 0) {
            if (t + 1 < ntiles) {
                const int kk = (t + 1) * BK;
                tdm_tile_load(A + (size_t)block_m * K + kk, ((t + 1) & 1) ? ldsA1 : ldsA0,
                              (unsigned)(CB - block_m), (unsigned)(K - kk), BM, (unsigned)K);
                tdm_tile_load(W + (size_t)block_n * K + kk, ((t + 1) & 1) ? ldsB1 : ldsB0,
                              (unsigned)(Nc - block_n), (unsigned)(K - kk), BN, (unsigned)K);
                __builtin_amdgcn_s_wait_tensorcnt(2);   // tile t done; 2 newest in flight
            } else {
                __builtin_amdgcn_s_wait_tensorcnt(0);
            }
        }
        __syncthreads();
#else
        const int cur = 0;
        const int kk = t * BK;
        {   // A tile: 128 rows x 4 x 16B chunks = 512 slots, 2 per thread
            #pragma unroll
            for (int s = 0; s < 2; ++s) {
                const int slot = tid + s * 256;
                const int r = slot >> 2, ch = slot & 3;
                const u32x4v* src = (const u32x4v*)(A + (size_t)(block_m + r) * K + kk) + ch;
                *(u32x4v*)&As[0][r][ch * 8] = *src;
                if (kk + BK < K) __builtin_prefetch((const void*)(src + (BK / 8)), 0, 3);
            }
        }
        {   // W tile: 64 rows x 4 chunks = 256 slots, 1 per thread, guarded
            const int r = tid >> 2, ch = tid & 3;
            if (block_n + r < Nc) {
                const u32x4v* src = (const u32x4v*)(W + (size_t)(block_n + r) * K + kk) + ch;
                *(u32x4v*)&Bs[0][r][ch * 8] = *src;
                if (kk + BK < K) __builtin_prefetch((const void*)(src + (BK / 8)), 0, 3);
            } else {
                u32x4v z = {0u, 0u, 0u, 0u};
                *(u32x4v*)&Bs[0][r][ch * 8] = z;
            }
        }
        __syncthreads();
#endif
        // ---- fragments (ISA 7.12.2 layouts); LDS rows 80B pitch, 16B aligned
        Frag a0, a1, b0, b1;
        {
            const u32x4v* r0 = (const u32x4v*)&As[cur][wm * 32 + lm][0];
            const u32x4v* r1 = (const u32x4v*)&As[cur][wm * 32 + 16 + lm][0];
            a0.q[0] = r0[halfsel];     a0.q[1] = r0[2 + halfsel];
            a1.q[0] = r1[halfsel];     a1.q[1] = r1[2 + halfsel];
            const u32x4v* c0 = (const u32x4v*)&Bs[cur][wn * 32 + lm][0];
            const u32x4v* c1 = (const u32x4v*)&Bs[cur][wn * 32 + 16 + lm][0];
            b0.q[0] = c0[halfsel];     b0.q[1] = c0[2 + halfsel];
            b1.q[0] = c1[halfsel];     b1.q[1] = c1[2 + halfsel];
        }
        acc00 = __builtin_amdgcn_wmma_f32_16x16x32_bf16(false, a0.v, false, b0.v, (short)0, acc00, false, false);
        acc01 = __builtin_amdgcn_wmma_f32_16x16x32_bf16(false, a0.v, false, b1.v, (short)0, acc01, false, false);
        acc10 = __builtin_amdgcn_wmma_f32_16x16x32_bf16(false, a1.v, false, b0.v, (short)0, acc10, false, false);
        acc11 = __builtin_amdgcn_wmma_f32_16x16x32_bf16(false, a1.v, false, b1.v, (short)0, acc11, false, false);
        __syncthreads();
    }

    // ---- epilogue + store
    float* Cf = (float*)Cout;
    unsigned short* Cb = (unsigned short*)Cout;
    #pragma unroll
    for (int fm = 0; fm < 2; ++fm) {
        #pragma unroll
        for (int fn = 0; fn < 2; ++fn) {
            const int ncol = block_n + wn * 32 + fn * 16 + lm;
            if (ncol >= Nc) continue;
            float bsum = 0.0f;
            if (bias)  bsum += bias[ncol];
            if (bias2) bsum += bias2[ncol];
            const v8f accv = (fm == 0) ? (fn == 0 ? acc00 : acc01)
                                       : (fn == 0 ? acc10 : acc11);
            #pragma unroll
            for (int v = 0; v < 8; ++v) {
                const size_t mrow = (size_t)(block_m + wm * 32 + fm * 16 + halfsel * 8 + v);
                float val = accv[v] + bsum;
                if (accumulate) val += Cf[mrow * ldc + ncol];
                if (epilogue == 1)      val = tanhf(val);
                else if (epilogue == 2) val = dev_sigmoid(val);
                if (out_bf16) Cb[mrow * ldc + ncol] = f2bf(val);
                else          Cf[mrow * ldc + ncol] = val;
            }
        }
    }
}

// ---------------------------------------------------------------------------
// fp32 -> bf16 bulk conversion (4 elems/thread)
// ---------------------------------------------------------------------------
__global__ __launch_bounds__(256)
void cvt_bf16_kernel(const float* __restrict__ src, unsigned short* __restrict__ dst, int n4)
{
    const int t = blockIdx.x * 256 + threadIdx.x;
    if (t >= n4) return;
    const float4 f = ((const float4*)src)[t];
    u16x4 o; o[0] = f2bf(f.x); o[1] = f2bf(f.y); o[2] = f2bf(f.z); o[3] = f2bf(f.w);
    ((u16x4*)dst)[t] = o;
}

// ---------------------------------------------------------------------------
// Block-wide reductions (256 threads)
// ---------------------------------------------------------------------------
__device__ __forceinline__ float blockReduceSum(float v, float* red, int tid) {
    red[tid] = v; __syncthreads();
    #pragma unroll
    for (int s = 128; s >= 1; s >>= 1) { if (tid < s) red[tid] += red[tid + s]; __syncthreads(); }
    float r = red[0]; __syncthreads(); return r;
}
__device__ __forceinline__ float blockReduceMax(float v, float* red, int tid) {
    red[tid] = v; __syncthreads();
    #pragma unroll
    for (int s = 128; s >= 1; s >>= 1) { if (tid < s) red[tid] = fmaxf(red[tid], red[tid + s]); __syncthreads(); }
    float r = red[0]; __syncthreads(); return r;
}

// ---------------------------------------------------------------------------
// LayerNorm + affine + ReLU over H=512; emits bf16 activation
// ---------------------------------------------------------------------------
__global__ __launch_bounds__(256)
void ln_relu_kernel(const float* __restrict__ z, const float* __restrict__ g,
                    const float* __restrict__ b, unsigned short* __restrict__ out)
{
    __shared__ float red[256];
    const int row = blockIdx.x, tid = threadIdx.x;
    const float x0 = z[(size_t)row * CH + tid];
    const float x1 = z[(size_t)row * CH + 256 + tid];
    const float mu = blockReduceSum(x0 + x1, red, tid) * (1.0f / CH);
    const float d0 = x0 - mu, d1 = x1 - mu;
    const float var = blockReduceSum(d0 * d0 + d1 * d1, red, tid) * (1.0f / CH);
    const float rs = rsqrtf(var + 1e-5f);
    out[(size_t)row * CH + tid]       = f2bf(fmaxf(d0 * rs * g[tid] + b[tid], 0.0f));
    out[(size_t)row * CH + 256 + tid] = f2bf(fmaxf(d1 * rs * g[256 + tid] + b[256 + tid], 0.0f));
}

// ---------------------------------------------------------------------------
// LSTM pointwise (gate order i,f,g,o); emits bf16 h, scatters into hr[:, :512]
// ---------------------------------------------------------------------------
__global__ __launch_bounds__(256)
void lstm_kernel(const float* __restrict__ z, const float* __restrict__ c0,
                 unsigned short* __restrict__ h, unsigned short* __restrict__ hr)
{
    const int t = blockIdx.x * 256 + threadIdx.x;     // < B*H
    const int bb = t >> 9, j = t & 511;
    const float* zr = z + (size_t)bb * (4 * CH);
    const float ig = dev_sigmoid(zr[j]);
    const float fg = dev_sigmoid(zr[CH + j]);
    const float gg = tanhf(zr[2 * CH + j]);
    const float og = dev_sigmoid(zr[3 * CH + j]);
    const float cc = fg * c0[t] + ig * gg;
    const unsigned short hh = f2bf(og * tanhf(cc));
    h[t] = hh;
    hr[(size_t)bb * (CH + CM) + j] = hh;
}

// ---------------------------------------------------------------------------
// Scatter previous read vector (bf16, or fp32 source for layer 0) into
// x[:, 256:320] (bf16, row stride 320)
// ---------------------------------------------------------------------------
__global__ __launch_bounds__(256)
void scatter_read_kernel(const float* __restrict__ rf32,
                         const unsigned short* __restrict__ rbf,
                         unsigned short* __restrict__ x, int use_f32)
{
    const int t = blockIdx.x * 256 + threadIdx.x;     // < B*M
    const int bb = t >> 6, m = t & 63;
    const unsigned short val = use_f32 ? f2bf(rf32[(size_t)bb * CM + m])
                                       : rbf[(size_t)bb * CM + m];
    x[(size_t)bb * (CV + CM) + CV + m] = val;
}

// ---------------------------------------------------------------------------
// Fused read head: content addressing + interpolate + shift + sharpen + read.
// One block (256 threads) per batch row; mem row (32 KB) staged in LDS.
// Emits bf16 r into rbuf and hr[:, 512:576].
// ---------------------------------------------------------------------------
__global__ __launch_bounds__(256)
void head_read_kernel(const float* __restrict__ o,        // [B, 70]
                      const float* __restrict__ mem,      // [B, N, M]
                      const float* __restrict__ wprev,    // [B, N]
                      unsigned short* __restrict__ rbuf,  // [B, M] bf16
                      unsigned short* __restrict__ hr)    // [B, H+M] bf16
{
    __shared__ float memrow[CN * CM];
    __shared__ float red[256];
    __shared__ float kv[CM];
    __shared__ float wsh[CN];
    __shared__ float scal[8];

    const int bb = blockIdx.x, tid = threadIdx.x;
    {
        const float4* src = (const float4*)(mem + (size_t)bb * CN * CM);
        float4* dst = (float4*)memrow;
        #pragma unroll
        for (int j = 0; j < 8; ++j) dst[tid + 256 * j] = src[tid + 256 * j];
    }
    const float* orow = o + (size_t)bb * (CM + 6);
    float kval = 0.0f;
    if (tid < CM) { kval = tanhf(orow[tid]); kv[tid] = kval; }
    __syncthreads();

    const float kn2 = blockReduceSum(tid < CM ? kval * kval : 0.0f, red, tid);
    if (tid == 0) {
        const float s0r = orow[CM + 2], s1r = orow[CM + 3], s2r = orow[CM + 4];
        const float mx = fmaxf(s0r, fmaxf(s1r, s2r));
        const float e0 = __expf(s0r - mx), e1 = __expf(s1r - mx), e2 = __expf(s2r - mx);
        const float es = e0 + e1 + e2;
        scal[0] = sqrtf(kn2);
        scal[1] = dev_softplus(orow[CM]);
        scal[2] = dev_sigmoid(orow[CM + 1]);
        scal[3] = e0 / es; scal[4] = e1 / es; scal[5] = e2 / es;
        scal[6] = 1.0f + dev_softplus(orow[CM + 5]);
    }
    __syncthreads();
    const float knorm = scal[0], beta = scal[1], gg = scal[2];
    const float s0 = scal[3], s1 = scal[4], s2 = scal[5], gamma = scal[6];

    float logit = -3.0e38f;
    if (tid < CN) {
        const float* row = &memrow[tid * CM];
        float dot = 0.0f, nn = 0.0f;
        #pragma unroll 8
        for (int m = 0; m < CM; ++m) { const float mv = row[m]; dot += mv * kv[m]; nn += mv * mv; }
        logit = beta * (dot / (sqrtf(nn) * knorm + 1e-16f));
    }
    const float mx = blockReduceMax(logit, red, tid);
    const float ex = (tid < CN) ? __expf(logit - mx) : 0.0f;
    const float se = blockReduceSum(ex, red, tid);
    if (tid < CN)
        wsh[tid] = gg * (ex / se) + (1.0f - gg) * wprev[(size_t)bb * CN + tid];
    __syncthreads();

    float wp = 0.0f;
    if (tid < CN) {
        const float wm1 = wsh[(tid + CN - 1) & (CN - 1)];
        const float w0  = wsh[tid];
        const float wp1 = wsh[(tid + 1) & (CN - 1)];
        wp = powf(s0 * wm1 + s1 * w0 + s2 * wp1, gamma);
    }
    const float sw = blockReduceSum(wp, red, tid);
    if (tid < CN) wsh[tid] = wp / (sw + 1e-16f);
    __syncthreads();

    if (tid < CM) {
        float acc = 0.0f;
        #pragma unroll 8
        for (int n = 0; n < CN; ++n) acc += wsh[n] * memrow[n * CM + tid];
        const unsigned short rb = f2bf(acc);
        rbuf[(size_t)bb * CM + tid] = rb;
        hr[(size_t)bb * (CH + CM) + CH + tid] = rb;
    }
}

// ---------------------------------------------------------------------------
// Host-side orchestration
// ---------------------------------------------------------------------------
static inline size_t align16(size_t x) { return (x + 15) & ~(size_t)15; }

extern "C" void kernel_launch(void* const* d_in, const int* in_sizes, int n_in,
                              void* d_out, int out_size, void* d_ws, size_t ws_size,
                              hipStream_t stream)
{
    const float* inputs = (const float*)d_in[0];
    const float* W1  = (const float*)d_in[1];
    const float* b1  = (const float*)d_in[2];
    const float* lng = (const float*)d_in[3];
    const float* lnb = (const float*)d_in[4];
    const float* W2  = (const float*)d_in[5];
    const float* b2  = (const float*)d_in[6];
    const float* Wih = (const float*)d_in[7];
    const float* Whh = (const float*)d_in[8];
    const float* bih = (const float*)d_in[9];
    const float* bhh = (const float*)d_in[10];
    const float* Wr  = (const float*)d_in[11];
    const float* br  = (const float*)d_in[12];
    // d_in[13] Ww, d_in[14] bw, d_in[20] ww0: dead (memory write never observed)
    const float* Wo  = (const float*)d_in[15];
    const float* bo  = (const float*)d_in[16];
    const float* mem0  = (const float*)d_in[17];
    const float* read0 = (const float*)d_in[18];
    const float* wr0   = (const float*)d_in[19];
    const float* h0 = (const float*)d_in[21];
    const float* c0 = (const float*)d_in[22];
    float* out = (float*)d_out;

    // ---- workspace layout ----
    char* cur = (char*)d_ws;
    float* z1;  float* zb;  float* orb;
    unsigned short *inpbf, *h0bf, *W1bf, *W2bf, *Wihbf, *Whhbf, *Wrbf, *Wobf;
    unsigned short *a1bf, *xbf, *hbf, *hrbf, *rbf;
    z1  = (float*)cur;           cur += align16((size_t)CB * CH * 4);
    zb  = (float*)cur;           cur += align16((size_t)CB * 4 * CH * 4);
    orb = (float*)cur;           cur += align16((size_t)CB * (CM + 6) * 4);
    inpbf = (unsigned short*)cur; cur += align16((size_t)CT * CB * CE * 2);
    h0bf  = (unsigned short*)cur; cur += align16((size_t)CT * CB * CH * 2);
    W1bf  = (unsigned short*)cur; cur += align16((size_t)CT * CH * CE * 2);
    W2bf  = (unsigned short*)cur; cur += align16((size_t)CT * CV * CH * 2);
    Wihbf = (unsigned short*)cur; cur += align16((size_t)CT * 4 * CH * (CV + CM) * 2);
    Whhbf = (unsigned short*)cur; cur += align16((size_t)CT * 4 * CH * CH * 2);
    Wrbf  = (unsigned short*)cur; cur += align16((size_t)CT * (CM + 6) * CH * 2);
    Wobf  = (unsigned short*)cur; cur += align16((size_t)CT * CE * (CH + CM) * 2);
    a1bf  = (unsigned short*)cur; cur += align16((size_t)CB * CH * 2);
    xbf   = (unsigned short*)cur; cur += align16((size_t)CB * (CV + CM) * 2);
    hbf   = (unsigned short*)cur; cur += align16((size_t)CB * CH * 2);
    hrbf  = (unsigned short*)cur; cur += align16((size_t)CB * (CH + CM) * 2);
    rbf   = (unsigned short*)cur; cur += align16((size_t)CB * CM * 2);

    const dim3 blk(256);

    // one-time fp32 -> bf16 conversions (all GEMM operands)
    {
        struct { const float* s; unsigned short* d; size_t n; } cv[] = {
            { inputs, inpbf, (size_t)CT * CB * CE },
            { h0,     h0bf,  (size_t)CT * CB * CH },
            { W1,     W1bf,  (size_t)CT * CH * CE },
            { W2,     W2bf,  (size_t)CT * CV * CH },
            { Wih,    Wihbf, (size_t)CT * 4 * CH * (CV + CM) },
            { Whh,    Whhbf, (size_t)CT * 4 * CH * CH },
            { Wr,     Wrbf,  (size_t)CT * (CM + 6) * CH },
            { Wo,     Wobf,  (size_t)CT * CE * (CH + CM) },
        };
        for (auto& c : cv) {
            const int n4 = (int)(c.n / 4);
            cvt_bf16_kernel<<<dim3((n4 + 255) / 256), blk, 0, stream>>>(c.s, c.d, n4);
        }
    }

    const int gy = CB / BM;   // 16

    for (int i = 0; i < CT; ++i) {
        // 1) z1 = inputs @ W1^T + b1                       [B,512] K=512
        gemm_bf16_wmma<<<dim3(CH / BN, gy), blk, 0, stream>>>(
            inpbf + (size_t)i * CB * CE, W1bf + (size_t)i * CH * CE,
            b1 + (size_t)i * CH, nullptr, z1, CH, CE, CH, 0, 0, 0);

        // 2) a1 = relu(LN(z1)*g + b)  -> bf16
        ln_relu_kernel<<<dim3(CB), blk, 0, stream>>>(
            z1, lng + (size_t)i * CH, lnb + (size_t)i * CH, a1bf);

        // 3) p = tanh(a1 @ W2^T + b2) -> x[:, :256] bf16   (ldc = 320)
        gemm_bf16_wmma<<<dim3(CV / BN, gy), blk, 0, stream>>>(
            a1bf, W2bf + (size_t)i * CV * CH, b2 + (size_t)i * CV, nullptr,
            xbf, CV, CH, CV + CM, 1, 0, 1);

        // 4) x[:, 256:320] = prev_read
        scatter_read_kernel<<<dim3(CB * CM / 256), blk, 0, stream>>>(
            read0 + (size_t)(CT - 1) * CB * CM, rbf, xbf, (i == 0) ? 1 : 0);

        // 5) z = x @ Wih^T + bih + bhh                     [B,2048] K=320
        gemm_bf16_wmma<<<dim3(4 * CH / BN, gy), blk, 0, stream>>>(
            xbf, Wihbf + (size_t)i * 4 * CH * (CV + CM),
            bih + (size_t)i * 4 * CH, bhh + (size_t)i * 4 * CH,
            zb, 4 * CH, CV + CM, 4 * CH, 0, 0, 0);

        // 6) z += h0 @ Whh^T                               [B,2048] K=512
        gemm_bf16_wmma<<<dim3(4 * CH / BN, gy), blk, 0, stream>>>(
            h0bf + (size_t)i * CB * CH, Whhbf + (size_t)i * 4 * CH * CH,
            nullptr, nullptr, zb, 4 * CH, CH, 4 * CH, 0, 1, 0);

        // 7) LSTM pointwise -> h (bf16), hr[:, :512]
        lstm_kernel<<<dim3(CB * CH / 256), blk, 0, stream>>>(
            zb, c0 + (size_t)i * CB * CH, hbf, hrbf);

        // 8) o_r = h @ Wr^T + br                           [B,70] K=512
        gemm_bf16_wmma<<<dim3((CM + 6 + BN - 1) / BN, gy), blk, 0, stream>>>(
            hbf, Wrbf + (size_t)i * (CM + 6) * CH, br + (size_t)i * (CM + 6), nullptr,
            orb, CM + 6, CH, CM + 6, 0, 0, 0);

        // 9) read head -> r (bf16) into rbf and hr[:, 512:576]
        head_read_kernel<<<dim3(CB), blk, 0, stream>>>(
            orb, mem0 + (size_t)i * CB * CN * CM, wr0 + (size_t)i * CB * CN, rbf, hrbf);

        // 10) out[i] = sigmoid(hr @ Wo^T + bo)             [B,512] K=576
        gemm_bf16_wmma<<<dim3(CE / BN, gy), blk, 0, stream>>>(
            hrbf, Wobf + (size_t)i * CE * (CH + CM), bo + (size_t)i * CE, nullptr,
            out + (size_t)i * CB * CE, CE, CH + CM, CE, 2, 0, 0);
    }
}
